// GCNNet_11647951307336
// MI455X (gfx1250) — compile-verified, matrix operand order checked by reading, not run on hardware
//
#include <hip/hip_runtime.h>

#define NN 100000
#define NE 1600000
#define DH 128
#define DOUT 40
#define DOUT_PAD 48

typedef __attribute__((ext_vector_type(16))) __bf16 v16bf;
typedef __attribute__((ext_vector_type(8)))  float  v8f;

__device__ __forceinline__ unsigned short f2bf(float f) {
  unsigned int u = __builtin_bit_cast(unsigned int, f);
  u += 0x7FFFu + ((u >> 16) & 1u);            // round-to-nearest-even
  return (unsigned short)(u >> 16);
}

// load 8 contiguous bf16 (16 bytes) into fragment elements [off, off+8)
__device__ __forceinline__ void load8bf(v16bf& f, int off, const unsigned short* p) {
  const uint4 w = *reinterpret_cast<const uint4*>(p);
  unsigned int ws[4] = {w.x, w.y, w.z, w.w};
#pragma unroll
  for (int i = 0; i < 4; ++i) {
    f[off + 2*i]     = __builtin_bit_cast(__bf16, (unsigned short)(ws[i] & 0xFFFFu));
    f[off + 2*i + 1] = __builtin_bit_cast(__bf16, (unsigned short)(ws[i] >> 16));
  }
}

// ---------------- degree / normalization ----------------
__global__ void deg_init(float* deg) {
  int i = blockIdx.x * blockDim.x + threadIdx.x;
  if (i < NN) deg[i] = 1.0f;                   // self-loop
}
__global__ void deg_count(const long long* __restrict__ dst, float* __restrict__ deg) {
  int e = blockIdx.x * blockDim.x + threadIdx.x;
  if (e < NE) atomicAdd(&deg[(int)dst[e]], 1.0f);
}
__global__ void dinv_kernel(const float* __restrict__ deg, float* __restrict__ dinv) {
  int i = blockIdx.x * blockDim.x + threadIdx.x;
  if (i < NN) dinv[i] = rsqrtf(deg[i]);
}

// ---------------- conversions ----------------
__global__ void cvt_bf16(const float* __restrict__ src, unsigned short* __restrict__ dst,
                         int n, int do_relu) {
  int i = blockIdx.x * blockDim.x + threadIdx.x;
  if (i >= n) return;
  float v = src[i];
  if (do_relu) v = fmaxf(v, 0.0f);
  dst[i] = f2bf(v);
}
// WT[n][k] = W[k][n] (bf16), zero-padded columns
__global__ void wt_transpose(const float* __restrict__ W, unsigned short* __restrict__ WT,
                             int d_out, int d_out_pad) {
  int i = blockIdx.x * blockDim.x + threadIdx.x;
  int total = d_out_pad * DH;
  if (i >= total) return;
  int n = i / DH, k = i % DH;
  float v = (n < d_out) ? W[(size_t)k * d_out + n] : 0.0f;
  WT[(size_t)n * DH + k] = f2bf(v);
}

// ---------------- WMMA GEMM: C[N x n_cols] = A[N x 128](bf16) * BT^T ----------------
// A row-major bf16 (ld=128); BT row-major bf16 [16*tiles_n x 128] = B^T.
__global__ void gemm_bf16_wmma(const unsigned short* __restrict__ A,
                               const unsigned short* __restrict__ BT,
                               float* __restrict__ C,
                               int tiles_n, int n_cols, int ldc) {
  int wid  = (int)((blockIdx.x * blockDim.x + threadIdx.x) >> 5);
  int lane = threadIdx.x & 31;
  int total_tiles = (NN / 16) * tiles_n;
  if (wid >= total_tiles) return;
  int tm = wid / tiles_n, tn = wid % tiles_n;
  int half = lane >> 4, mr = lane & 15;

  const unsigned short* arow = A  + (size_t)(tm * 16 + mr) * DH;
  const unsigned short* brow = BT + (size_t)(tn * 16 + mr) * DH;

  v8f acc = {0.f, 0.f, 0.f, 0.f, 0.f, 0.f, 0.f, 0.f};
#pragma unroll
  for (int k0 = 0; k0 < DH; k0 += 32) {
    v16bf fa, fb;
    // A fragment: lane = row (M), two contiguous 16B chunks per K-step
    load8bf(fa, 0, arow + k0 +      half * 8);
    load8bf(fa, 8, arow + k0 + 16 + half * 8);
    // B fragment: lane = col (N), identical K striping over W^T rows
    load8bf(fb, 0, brow + k0 +      half * 8);
    load8bf(fb, 8, brow + k0 + 16 + half * 8);
    acc = __builtin_amdgcn_wmma_f32_16x16x32_bf16(false, fa, false, fb,
                                                  (short)0, acc, false, false);
  }
  // C layout: lane holds column n = lane%16; VGPR r -> row r + 8*(lane/16)
  int col = tn * 16 + mr;
  if (col < n_cols) {
    float* crow = C + (size_t)(tm * 16 + 8 * half) * ldc + col;
#pragma unroll
    for (int r = 0; r < 8; ++r) crow[(size_t)r * ldc] = acc[r];
  }
}

// ---------------- aggregation ----------------
// a[i][c] = bias[c] + h[i][c] * dinv[i]^2   (self-loop contribution + bias)
__global__ void agg_init(const float* __restrict__ h, const float* __restrict__ dinv,
                         const float* __restrict__ bias, float* __restrict__ a,
                         int d, int ldh) {
  int i = blockIdx.x * blockDim.x + threadIdx.x;
  int total = NN * d;
  if (i >= total) return;
  int node = i / d, c = i % d;
  float di = dinv[node];
  a[i] = bias[c] + h[(size_t)node * ldh + c] * di * di;
}
// one wave per edge; lane moves float4 (32*16B = 512B = full 128-dim row)
__global__ void agg_edges128(const long long* __restrict__ src, const long long* __restrict__ dst,
                             const float* __restrict__ dinv, const float* __restrict__ h,
                             float* __restrict__ a) {
  int w = (int)((blockIdx.x * blockDim.x + threadIdx.x) >> 5);
  int lane = threadIdx.x & 31;
  if (w >= NE) return;
  int s = (int)src[w], t = (int)dst[w];
  float nrm = dinv[s] * dinv[t];
  const float4 v = *reinterpret_cast<const float4*>(h + (size_t)s * DH + lane * 4);
  float* o = a + (size_t)t * DH + lane * 4;
  atomicAdd(o + 0, v.x * nrm);
  atomicAdd(o + 1, v.y * nrm);
  atomicAdd(o + 2, v.z * nrm);
  atomicAdd(o + 3, v.w * nrm);
}
__global__ void agg_edges40(const long long* __restrict__ src, const long long* __restrict__ dst,
                            const float* __restrict__ dinv, const float* __restrict__ h,
                            int ldh, float* __restrict__ a) {
  int w = (int)((blockIdx.x * blockDim.x + threadIdx.x) >> 5);
  int lane = threadIdx.x & 31;
  if (w >= NE) return;
  int s = (int)src[w], t = (int)dst[w];
  float nrm = dinv[s] * dinv[t];
  for (int d = lane; d < DOUT; d += 32)
    atomicAdd(&a[(size_t)t * DOUT + d], h[(size_t)s * ldh + d] * nrm);
}

// ---------------- in-place log_softmax over 40 cols, one wave32 per node ----------------
__global__ void log_softmax40(float* __restrict__ out) {
  int w = (int)((blockIdx.x * blockDim.x + threadIdx.x) >> 5);
  int lane = threadIdx.x & 31;
  if (w >= NN) return;
  float* row = out + (size_t)w * DOUT;
  float x0 = row[lane];
  float x1 = (lane < DOUT - 32) ? row[32 + lane] : -3.402823e38f;
  float m = fmaxf(x0, x1);
#pragma unroll
  for (int off = 16; off > 0; off >>= 1) m = fmaxf(m, __shfl_xor(m, off, 32));
  float s = __expf(x0 - m) + ((lane < DOUT - 32) ? __expf(x1 - m) : 0.0f);
#pragma unroll
  for (int off = 16; off > 0; off >>= 1) s += __shfl_xor(s, off, 32);
  float lse = m + __logf(s);
  row[lane] = x0 - lse;
  if (lane < DOUT - 32) row[32 + lane] = x1 - lse;
}

extern "C" void kernel_launch(void* const* d_in, const int* in_sizes, int n_in,
                              void* d_out, int out_size, void* d_ws, size_t ws_size,
                              hipStream_t stream) {
  (void)in_sizes; (void)n_in; (void)out_size; (void)ws_size;
  const float*     x   = (const float*)d_in[0];
  const long long* ei  = (const long long*)d_in[1];
  const float*     W1  = (const float*)d_in[2];
  const float*     b1  = (const float*)d_in[3];
  const float*     W2  = (const float*)d_in[4];
  const float*     b2  = (const float*)d_in[5];
  const float*     W3  = (const float*)d_in[6];
  const float*     b3  = (const float*)d_in[7];
  const long long* src = ei;
  const long long* dst = ei + NE;
  float* out = (float*)d_out;

  char* ws = (char*)d_ws;
  size_t off = 0;
  auto alloc = [&](size_t bytes) -> void* {
    void* p = ws + off;
    off += (bytes + 255) & ~(size_t)255;
    return p;
  };
  float*          deg  = (float*)alloc((size_t)NN * 4);
  float*          dinv = (float*)alloc((size_t)NN * 4);
  unsigned short* xb   = (unsigned short*)alloc((size_t)NN * DH * 2);  // bf16 activations
  float*          h    = (float*)alloc((size_t)NN * DH * 4);           // GEMM out (also N x 48)
  float*          a    = (float*)alloc((size_t)NN * DH * 4);           // aggregation accum
  unsigned short* wt1  = (unsigned short*)alloc((size_t)DH * DH * 2);
  unsigned short* wt2  = (unsigned short*)alloc((size_t)DH * DH * 2);
  unsigned short* wt3  = (unsigned short*)alloc((size_t)DOUT_PAD * DH * 2);

  const int TB = 256;
  const int nblk_node  = (NN + TB - 1) / TB;
  const int nblk_edge  = (NE + TB - 1) / TB;
  const int nblk_feat  = (NN * DH + TB - 1) / TB;
  const int nblk_ewave = (int)(((size_t)NE * 32 + TB - 1) / TB);

  // --- normalization coefficients ---
  deg_init  <<<nblk_node, TB, 0, stream>>>(deg);
  deg_count <<<nblk_edge, TB, 0, stream>>>(dst, deg);
  dinv_kernel<<<nblk_node, TB, 0, stream>>>(deg, dinv);

  // --- weight prep (bf16, transposed, padded) ---
  wt_transpose<<<(DH * DH + TB - 1) / TB, TB, 0, stream>>>(W1, wt1, DH, DH);
  wt_transpose<<<(DH * DH + TB - 1) / TB, TB, 0, stream>>>(W2, wt2, DH, DH);
  wt_transpose<<<(DOUT_PAD * DH + TB - 1) / TB, TB, 0, stream>>>(W3, wt3, DOUT, DOUT_PAD);

  // --- layer 1 ---
  cvt_bf16<<<nblk_feat, TB, 0, stream>>>(x, xb, NN * DH, 0);
  gemm_bf16_wmma<<<(NN / 16) * 8 / 8, TB, 0, stream>>>(xb, wt1, h, 8, DH, DH);
  agg_init<<<nblk_feat, TB, 0, stream>>>(h, dinv, b1, a, DH, DH);
  agg_edges128<<<nblk_ewave, TB, 0, stream>>>(src, dst, dinv, h, a);
  cvt_bf16<<<nblk_feat, TB, 0, stream>>>(a, xb, NN * DH, 1);  // ReLU + bf16

  // --- layer 2 ---
  gemm_bf16_wmma<<<(NN / 16) * 8 / 8, TB, 0, stream>>>(xb, wt2, h, 8, DH, DH);
  agg_init<<<nblk_feat, TB, 0, stream>>>(h, dinv, b2, a, DH, DH);
  agg_edges128<<<nblk_ewave, TB, 0, stream>>>(src, dst, dinv, h, a);
  cvt_bf16<<<nblk_feat, TB, 0, stream>>>(a, xb, NN * DH, 1);

  // --- layer 3 (40 cols padded to 48) ---
  {
    int tiles = (NN / 16) * 3;                         // 18750 waves
    int blocks = (tiles * 32 + TB - 1) / TB;
    gemm_bf16_wmma<<<blocks, TB, 0, stream>>>(xb, wt3, h, 3, DOUT, DOUT_PAD);
  }
  agg_init<<<(NN * DOUT + TB - 1) / TB, TB, 0, stream>>>(h, dinv, b3, out, DOUT, DOUT_PAD);
  agg_edges40<<<nblk_ewave, TB, 0, stream>>>(src, dst, dinv, h, DOUT_PAD, out);
  log_softmax40<<<(int)(((size_t)NN * 32 + TB - 1) / TB), TB, 0, stream>>>(out);
}
